// MultiHeadSelfAttention_15839839388317
// MI455X (gfx1250) — matrix operand order repaired and checked by reading, not compiled, not run on hardware
//
#include <hip/hip_runtime.h>
#include <hip/hip_bf16.h>

typedef __attribute__((ext_vector_type(16))) __bf16 v16bf;
typedef __attribute__((ext_vector_type(8)))  float  v8f;

#define EMBED 1024
#define NHEAD 16
#define DHEAD 64
#define BATCH 4
#define SEQ   2048
#define MROWS (BATCH * SEQ)   /* 8192 */
#define HD    (NHEAD * DHEAD) /* 1024 */

__device__ __forceinline__ unsigned short f2bf(float f) {
  unsigned int u = __float_as_uint(f);
  u += 0x7FFFu + ((u >> 16) & 1u);           // round-to-nearest-even
  return (unsigned short)(u >> 16);
}

union FragU {
  v16bf v;
  uint4 q[2];
};

// A-side fragment (16x32 bf16): lane row = lane&15, kb = 16*(lane>>4);
// halves 0-15 = K[kb..kb+15] (32 contiguous bytes).
__device__ __forceinline__ v16bf load_a_frag(const unsigned short* p) {
  FragU f;
  f.q[0] = *reinterpret_cast<const uint4*>(p);
  f.q[1] = *reinterpret_cast<const uint4*>(p + 8);
  return f.v;
}

// B-side fragment (32x16 bf16): lane col = lane&15, kb = 16*(lane>>4);
// halves 0-15 = K[kb..kb+15] (32 contiguous bytes).
__device__ __forceinline__ v16bf load_b_frag(const unsigned short* p) {
  FragU f;
  f.q[0] = *reinterpret_cast<const uint4*>(p);
  f.q[1] = *reinterpret_cast<const uint4*>(p + 8);
  return f.v;
}

// ---------------------------------------------------------------- converts
__global__ __launch_bounds__(256) void k_f32_to_bf16(const float* __restrict__ in,
                                                     unsigned short* __restrict__ out,
                                                     int n) {
  int i = blockIdx.x * 256 + threadIdx.x;
  if (i < n) out[i] = f2bf(in[i]);
}

// in: fp32 [R][C] row-major -> out: bf16 [C][R]
__global__ __launch_bounds__(256) void k_transpose_bf16(const float* __restrict__ in,
                                                        unsigned short* __restrict__ out,
                                                        int R, int C) {
  int i = blockIdx.x * 256 + threadIdx.x;
  int r = i / C, c = i - r * C;
  out[(size_t)c * R + r] = f2bf(in[i]);
}

// ---------------------------------------------------------------- GEMM
// C[M][N] = A[M][K] * Bt[N][K]^T + bias[N], K = N = 1024.
// Block: 256 threads = 8 waves in a 4(M) x 2(N) grid; wave tile = 32x64.
// mode 0: bf16 out at [b][h][s][d]   (Q, K)
// mode 1: bf16 out at [b][h][d][s]   (V transposed for attention B-frags)
// mode 2: fp32 out row-major [M][HD] (final projection)
__global__ __launch_bounds__(256) void k_gemm_wmma(const unsigned short* __restrict__ A,
                                                   const unsigned short* __restrict__ Bt,
                                                   const float* __restrict__ bias,
                                                   void* __restrict__ out, int mode) {
  constexpr int K = 1024;
  const int lane = threadIdx.x & 31;
  const int wid  = threadIdx.x >> 5;
  const int wm = wid & 3, wn = wid >> 2;
  const int row0 = blockIdx.x * 128 + wm * 32;
  const int col0 = blockIdx.y * 128 + wn * 64;
  const int lr  = lane & 15;
  const int kba = (lane >> 4) * 16;
  const int kbb = (lane >> 4) * 16;

  v8f acc00 = {}, acc01 = {}, acc02 = {}, acc03 = {};
  v8f acc10 = {}, acc11 = {}, acc12 = {}, acc13 = {};

  const unsigned short* pa0 = A + (size_t)(row0 + lr) * K + kba;
  const unsigned short* pa1 = A + (size_t)(row0 + 16 + lr) * K + kba;
  const unsigned short* pb0 = Bt + (size_t)(col0 + 0 * 16 + lr) * K + kbb;
  const unsigned short* pb1 = Bt + (size_t)(col0 + 1 * 16 + lr) * K + kbb;
  const unsigned short* pb2 = Bt + (size_t)(col0 + 2 * 16 + lr) * K + kbb;
  const unsigned short* pb3 = Bt + (size_t)(col0 + 3 * 16 + lr) * K + kbb;

  for (int k0 = 0; k0 < K; k0 += 32) {
    v16bf a0 = load_a_frag(pa0 + k0);
    v16bf a1 = load_a_frag(pa1 + k0);
    v16bf b0 = load_b_frag(pb0 + k0);
    v16bf b1 = load_b_frag(pb1 + k0);
    v16bf b2 = load_b_frag(pb2 + k0);
    v16bf b3 = load_b_frag(pb3 + k0);
    if (k0 + 32 < K) {
      __builtin_prefetch(pa0 + k0 + 32, 0, 0);   // global_prefetch_b8
      __builtin_prefetch(pb0 + k0 + 32, 0, 0);
    }
    acc00 = __builtin_amdgcn_wmma_f32_16x16x32_bf16(false, a0, false, b0, (short)0, acc00, false, false);
    acc01 = __builtin_amdgcn_wmma_f32_16x16x32_bf16(false, a0, false, b1, (short)0, acc01, false, false);
    acc02 = __builtin_amdgcn_wmma_f32_16x16x32_bf16(false, a0, false, b2, (short)0, acc02, false, false);
    acc03 = __builtin_amdgcn_wmma_f32_16x16x32_bf16(false, a0, false, b3, (short)0, acc03, false, false);
    acc10 = __builtin_amdgcn_wmma_f32_16x16x32_bf16(false, a1, false, b0, (short)0, acc10, false, false);
    acc11 = __builtin_amdgcn_wmma_f32_16x16x32_bf16(false, a1, false, b1, (short)0, acc11, false, false);
    acc12 = __builtin_amdgcn_wmma_f32_16x16x32_bf16(false, a1, false, b2, (short)0, acc12, false, false);
    acc13 = __builtin_amdgcn_wmma_f32_16x16x32_bf16(false, a1, false, b3, (short)0, acc13, false, false);
  }

  // C/D layout: lane holds col n = col0 + nt*16 + (lane&15); VGPR r holds
  // row m = row0 + mt*16 + r + 8*(lane>>4).
  v8f accs[2][4] = {{acc00, acc01, acc02, acc03}, {acc10, acc11, acc12, acc13}};
  const int rbase = (lane >> 4) * 8;
#pragma unroll
  for (int nt = 0; nt < 4; ++nt) {
    const int n = col0 + nt * 16 + lr;
    const float bv = bias[n];
#pragma unroll
    for (int mt = 0; mt < 2; ++mt) {
#pragma unroll
      for (int r = 0; r < 8; ++r) {
        const int m = row0 + mt * 16 + rbase + r;
        const float val = accs[mt][nt][r] + bv;
        if (mode == 2) {
          ((float*)out)[(size_t)m * HD + n] = val;
        } else {
          const int b = m >> 11, s = m & (SEQ - 1);
          const int h = n >> 6, d = n & 63;
          const size_t idx =
              (mode == 0) ? ((((size_t)(b * NHEAD + h)) * SEQ + s) * DHEAD + d)
                          : ((((size_t)(b * NHEAD + h)) * DHEAD + d) * SEQ + s);
          ((unsigned short*)out)[idx] = f2bf(val);
        }
      }
    }
  }
}

// ---------------------------------------------------------------- attention
// One wave per 16 q-rows; flash attention over key blocks of 32.
// Q,K: bf16 [b][h][s][d]; Vt: bf16 [b][h][d][s]; ctx: bf16 [b*S][H*D].
__global__ __launch_bounds__(256) void k_attention(const unsigned short* __restrict__ Q,
                                                   const unsigned short* __restrict__ Kd,
                                                   const unsigned short* __restrict__ Vt,
                                                   unsigned short* __restrict__ ctx) {
  __shared__ unsigned short lds[8][16 * 32];   // per-wave 16x32 bf16 P tile
  const int lane = threadIdx.x & 31;
  const int wid  = threadIdx.x >> 5;
  const int bh   = blockIdx.x;          // b*NHEAD + h
  const int b    = bh >> 4;
  const int h    = bh & 15;
  const int q0   = blockIdx.y * 128 + wid * 16;
  const int lr   = lane & 15;
  const int kba  = (lane >> 4) * 16;
  const int kbb  = (lane >> 4) * 16;

  const unsigned short* Qh = Q  + (size_t)bh * SEQ * DHEAD;
  const unsigned short* Kh = Kd + (size_t)bh * SEQ * DHEAD;
  const unsigned short* Vh = Vt + (size_t)bh * DHEAD * SEQ;

  const v16bf qa0 = load_a_frag(Qh + (size_t)(q0 + lr) * DHEAD + kba);
  const v16bf qa1 = load_a_frag(Qh + (size_t)(q0 + lr) * DHEAD + 32 + kba);

  float mrow[8], lrow[8];
  v8f acc[4] = {};
  v8f zero = {};
#pragma unroll
  for (int r = 0; r < 8; ++r) { mrow[r] = -3.0e38f; lrow[r] = 0.0f; }

  unsigned short* myLds = &lds[wid][0];
  const int rbase = (lane >> 4) * 8;

  for (int kt = 0; kt < SEQ; kt += 32) {
    // --- scores: 16(q) x 32(keys), two 16-col WMMA tiles, K-dim = d (64)
    v8f sc[2];
#pragma unroll
    for (int ct = 0; ct < 2; ++ct) {
      const unsigned short* kp = Kh + (size_t)(kt + ct * 16 + lr) * DHEAD;
      v16bf kb0 = load_b_frag(kp + kbb);
      v16bf kb1 = load_b_frag(kp + 32 + kbb);
      v8f t = __builtin_amdgcn_wmma_f32_16x16x32_bf16(false, qa0, false, kb0, (short)0, zero, false, false);
      t = __builtin_amdgcn_wmma_f32_16x16x32_bf16(false, qa1, false, kb1, (short)0, t, false, false);
      sc[ct] = t;
    }
    // --- online softmax: row reductions across 16 lanes of each half-wave
    float alpha[8];
#pragma unroll
    for (int r = 0; r < 8; ++r) {
      float mx = fmaxf(sc[0][r], sc[1][r]);
      mx = fmaxf(mx, __shfl_xor(mx, 1));
      mx = fmaxf(mx, __shfl_xor(mx, 2));
      mx = fmaxf(mx, __shfl_xor(mx, 4));
      mx = fmaxf(mx, __shfl_xor(mx, 8));
      const float mnew = fmaxf(mrow[r], mx);
      const float p0 = __expf(sc[0][r] - mnew);
      const float p1 = __expf(sc[1][r] - mnew);
      sc[0][r] = p0;
      sc[1][r] = p1;
      float s = p0 + p1;
      s += __shfl_xor(s, 1);
      s += __shfl_xor(s, 2);
      s += __shfl_xor(s, 4);
      s += __shfl_xor(s, 8);
      alpha[r] = __expf(mrow[r] - mnew);
      lrow[r] = lrow[r] * alpha[r] + s;
      mrow[r] = mnew;
    }
#pragma unroll
    for (int nt = 0; nt < 4; ++nt)
#pragma unroll
      for (int r = 0; r < 8; ++r) acc[nt][r] *= alpha[r];

    // --- transpose P (C-layout -> A-layout) through per-wave LDS
#pragma unroll
    for (int r = 0; r < 8; ++r) {
      myLds[(rbase + r) * 32 + lr]      = f2bf(sc[0][r]);
      myLds[(rbase + r) * 32 + 16 + lr] = f2bf(sc[1][r]);
    }
    asm volatile("s_wait_dscnt 0" ::: "memory");   // wave-local LDS RAW fence
    const v16bf pa = load_a_frag(myLds + lr * 32 + kba);

    // --- O += P(16x32) x V(32x16) for the four d-tiles
#pragma unroll
    for (int nt = 0; nt < 4; ++nt) {
      const unsigned short* vp = Vh + (size_t)(nt * 16 + lr) * SEQ + kt + kbb;
      v16bf vb = load_b_frag(vp);
      acc[nt] = __builtin_amdgcn_wmma_f32_16x16x32_bf16(false, pa, false, vb, (short)0, acc[nt], false, false);
    }
  }

  // --- normalize and emit context (bf16 row-major [b*S][H*D])
#pragma unroll
  for (int nt = 0; nt < 4; ++nt) {
#pragma unroll
    for (int r = 0; r < 8; ++r) {
      const size_t m = (size_t)b * SEQ + q0 + rbase + r;
      const int n = h * DHEAD + nt * 16 + lr;
      ctx[m * HD + n] = f2bf(acc[nt][r] / lrow[r]);
    }
  }
}

// ---------------------------------------------------------------- launch
extern "C" void kernel_launch(void* const* d_in, const int* in_sizes, int n_in,
                              void* d_out, int out_size, void* d_ws, size_t ws_size,
                              hipStream_t stream) {
  (void)in_sizes; (void)n_in; (void)out_size; (void)ws_size;
  const float* x  = (const float*)d_in[0];
  const float* WQ = (const float*)d_in[1];
  const float* bQ = (const float*)d_in[2];
  const float* WK = (const float*)d_in[3];
  const float* bK = (const float*)d_in[4];
  const float* WV = (const float*)d_in[5];
  const float* bV = (const float*)d_in[6];
  const float* WO = (const float*)d_in[7];
  const float* bO = (const float*)d_in[8];

  char* ws = (char*)d_ws;
  size_t off = 0;
  auto carve = [&](size_t bytes) -> void* {
    void* p = ws + off;
    off += (bytes + 255) & ~(size_t)255;
    return p;
  };
  unsigned short* xb  = (unsigned short*)carve((size_t)MROWS * EMBED * 2); // 16 MB
  unsigned short* wqt = (unsigned short*)carve((size_t)HD * EMBED * 2);    //  2 MB
  unsigned short* wkt = (unsigned short*)carve((size_t)HD * EMBED * 2);
  unsigned short* wvt = (unsigned short*)carve((size_t)HD * EMBED * 2);
  unsigned short* wot = (unsigned short*)carve((size_t)EMBED * HD * 2);
  unsigned short* Qb  = (unsigned short*)carve((size_t)MROWS * HD * 2);    // 16 MB
  unsigned short* Kb  = (unsigned short*)carve((size_t)MROWS * HD * 2);
  unsigned short* Vtb = (unsigned short*)carve((size_t)MROWS * HD * 2);
  unsigned short* ctx = (unsigned short*)carve((size_t)MROWS * HD * 2);

  // 1) precision conversion / weight transposes
  k_f32_to_bf16<<<(MROWS * EMBED) / 256, 256, 0, stream>>>(x, xb, MROWS * EMBED);
  k_transpose_bf16<<<(EMBED * HD) / 256, 256, 0, stream>>>(WQ, wqt, EMBED, HD);
  k_transpose_bf16<<<(EMBED * HD) / 256, 256, 0, stream>>>(WK, wkt, EMBED, HD);
  k_transpose_bf16<<<(EMBED * HD) / 256, 256, 0, stream>>>(WV, wvt, EMBED, HD);
  k_transpose_bf16<<<(HD * EMBED) / 256, 256, 0, stream>>>(WO, wot, HD, EMBED);

  // 2) QKV projections (WMMA)
  dim3 gg(MROWS / 128, HD / 128);
  k_gemm_wmma<<<gg, 256, 0, stream>>>(xb, wqt, bQ, Qb, 0);
  k_gemm_wmma<<<gg, 256, 0, stream>>>(xb, wkt, bK, Kb, 0);
  k_gemm_wmma<<<gg, 256, 0, stream>>>(xb, wvt, bV, Vtb, 1);

  // 3) flash attention (WMMA), no 1/sqrt(d) scale per reference
  dim3 ga(BATCH * NHEAD, SEQ / 128);
  k_attention<<<ga, 256, 0, stream>>>(Qb, Kb, Vtb, ctx);

  // 4) output projection -> fp32 d_out
  k_gemm_wmma<<<gg, 256, 0, stream>>>(ctx, wot, bO, d_out, 2);
}
